// GQA_38568806318386
// MI455X (gfx1250) — compile-verified, matrix operand order checked by reading, not run on hardware
//
#include <hip/hip_runtime.h>
#include <type_traits>

#define DM  2048   // d_model
#define NH  32     // query heads
#define NKV 8      // kv heads
#define HD  64     // head dim
#define BB  2      // batch
#define TT  2048   // seq len

typedef _Float16 h8   __attribute__((ext_vector_type(8)));
typedef _Float16 v16h __attribute__((ext_vector_type(16)));
typedef float    v8f  __attribute__((ext_vector_type(8)));

union H16 { v16h v; h8 h[2]; };

// Load a 16x32 f16 WMMA fragment from a row-major matrix `p` (leading dim ld).
// A-fragment: row0 = M tile base; B-fragment (for A·Wᵀ): row0 = N tile base
// (B column n == W row n, contiguous along K).  ISA 7.12.2 layout:
// lane<16 holds K {k0..k0+7, k0+16..k0+23}; lane>=16 the +8 halves.
__device__ __forceinline__ v16h load_frag(const _Float16* __restrict__ p, int ld,
                                          int row0, int k0, int lane) {
  const int r  = row0 + (lane & 15);
  const int kb = k0 + ((lane >> 4) << 3);
  const _Float16* base = p + (size_t)r * ld + kb;
  H16 u;
  u.h[0] = *(const h8*)(base);
  u.h[1] = *(const h8*)(base + 16);
  return u.v;
}

__device__ __forceinline__ v8f wmma_f16(v16h a, v16h b, v8f c) {
  return __builtin_amdgcn_wmma_f32_16x16x32_f16(false, a, false, b, (short)0, c,
                                                false, false);
}

// Cross-half (xor-16) exchange as a pure VALU op; avoids the LDS pipeline
// (ds_bpermute + s_wait_dscnt) in the softmax serial chain.
__device__ __forceinline__ float xhalf_swap(float v) {
#if __has_builtin(__builtin_amdgcn_permlanex16)
  int i = __builtin_bit_cast(int, v);
  int o = __builtin_amdgcn_permlanex16(i, i, 0x76543210, (int)0xFEDCBA98,
                                       false, false);
  return __builtin_bit_cast(float, o);
#else
  return __shfl_xor(v, 16, 32);
#endif
}

__global__ void cvt_f32_f16(const float* __restrict__ in,
                            _Float16* __restrict__ out, int n) {
  int i = blockIdx.x * blockDim.x + threadIdx.x;
  if (i < n) out[i] = (_Float16)in[i];
}

// Y = A (MxK, f16) · W(NxK, f16)^T, f32 accumulate.
// 8 waves/block; wave tile 32(M)x64(N); block tile 128x128.
// MODE 0: f32 row-major out (final projection)
// MODE 1: RoPE + 1/sqrt(D) pre-scale + pack f16 Q -> (B,NH,T,HD)
// MODE 2: RoPE + pack f16 K -> (B,NKV,T,HD)
// MODE 3: pack f16 V transposed -> (B,NKV,HD,T)
template <int MODE>
__global__ void gemm_wmma(const _Float16* __restrict__ A,
                          const _Float16* __restrict__ W,
                          float* __restrict__ outF, _Float16* __restrict__ outH,
                          const float* __restrict__ cosp,
                          const float* __restrict__ sinp,
                          int M, int N, int K) {
  const int lane = threadIdx.x & 31;
  const int wave = threadIdx.x >> 5;
  const int wm = wave & 3;
  const int wn = wave >> 2;
  const int m0 = blockIdx.y * 128 + wm * 32;
  const int n0 = blockIdx.x * 128 + wn * 64;   // 64-aligned => spans one head

  v8f acc[2][4] = {};

  for (int k0 = 0; k0 < K; k0 += 32) {
    if (k0 + 64 < K)  // speculative prefetch of the A stream
      __builtin_prefetch(A + (size_t)(m0 + (lane & 15)) * K + k0 + 64, 0, 1);
    v16h a0 = load_frag(A, K, m0, k0, lane);
    v16h a1 = load_frag(A, K, m0 + 16, k0, lane);
#pragma unroll
    for (int j = 0; j < 4; ++j) {
      v16h b = load_frag(W, K, n0 + 16 * j, k0, lane);
      acc[0][j] = wmma_f16(a0, b, acc[0][j]);
      acc[1][j] = wmma_f16(a1, b, acc[1][j]);
    }
  }

  const int off8 = (lane >> 4) << 3;
  const int c0 = lane & 15;

  if constexpr (MODE == 0) {
#pragma unroll
    for (int i = 0; i < 2; ++i)
#pragma unroll
      for (int j = 0; j < 4; ++j) {
        const int col = n0 + 16 * j + c0;
#pragma unroll
        for (int r = 0; r < 8; ++r) {
          const int row = m0 + 16 * i + off8 + r;
          outF[(size_t)row * N + col] = acc[i][j][r];
        }
      }
  } else if constexpr (MODE == 1 || MODE == 2) {
    constexpr int HEADS = (MODE == 1) ? NH : NKV;
    // Fold the attention 1/sqrt(HD) into Q (exact: power of two).
    constexpr float POST = (MODE == 1) ? 0.125f : 1.0f;
#pragma unroll
    for (int i = 0; i < 2; ++i)
#pragma unroll
      for (int j = 0; j < 4; ++j) {
        const int col = n0 + 16 * j + c0;
        const int hh = col >> 6;
        const int d = col & 63;
        const float sgn = (d < 32) ? -1.0f : 1.0f;
        const int jp = j ^ 2;  // accumulator holding element at d +/- 32
#pragma unroll
        for (int r = 0; r < 8; ++r) {
          const int row = m0 + 16 * i + off8 + r;
          const int b = row / TT;
          const int t = row % TT;
          const float u = acc[i][j][r];
          const float rot = sgn * acc[i][jp][r];
          const float cv = cosp[t * HD + d];
          const float sv = sinp[t * HD + d];
          outH[(((size_t)b * HEADS + hh) * TT + t) * HD + d] =
              (_Float16)((u * cv + rot * sv) * POST);
        }
      }
  } else {  // MODE 3: V transposed, packed 16B stores along t
#pragma unroll
    for (int i = 0; i < 2; ++i) {
      const int rowb = m0 + 16 * i + off8;
      const int b = rowb / TT;
      const int t0 = rowb % TT;
#pragma unroll
      for (int j = 0; j < 4; ++j) {
        const int col = n0 + 16 * j + c0;
        const int kv = col >> 6;
        const int d = col & 63;
        h8 pk;
#pragma unroll
        for (int r = 0; r < 8; ++r) pk[r] = (_Float16)acc[i][j][r];
        *(h8*)(outH + (((size_t)b * NKV + kv) * HD + d) * TT + t0) = pk;
      }
    }
  }
}

// Causal flash attention, fully transposed:  Sᵀ = K·Qᵀ,  Oᵀ = Vᵀ·Pᵀ.
// grid = (T/128, B*NH), block = 256 (8 waves); each wave owns 16 queries.
// One query per lane (lane&15) => softmax stats, rescale, and 1/li are
// in-lane scalars; Pᵀ C-fragments are directly valid B-fragments (no LDS).
// Key loop is split: full (unmasked) chunks carry zero mask arithmetic;
// only the diagonal boundary chunks pay for causal compares.
__global__ void attn_wmma(const _Float16* __restrict__ qh,
                          const _Float16* __restrict__ kh,
                          const _Float16* __restrict__ vth,
                          _Float16* __restrict__ yh) {
  const int lane = threadIdx.x & 31;
  const int wave = threadIdx.x >> 5;
  const int bh = blockIdx.y;
  const int b = bh >> 5;        // / NH
  const int h = bh & 31;        // % NH
  const int kv = h >> 2;        // / (NH/NKV)
  const int m0 = blockIdx.x * 128 + wave * 16;

  const _Float16* qb = qh + (size_t)(b * NH + h) * TT * HD;
  const _Float16* kb = kh + (size_t)(b * NKV + kv) * TT * HD;
  const _Float16* vb = vth + (size_t)(b * NKV + kv) * HD * TT;

  // B-fragments of Qᵀ (columns = queries; Q carries the 1/8 scale already).
  const v16h bq0 = load_frag(qb, HD, m0, 0, lane);
  const v16h bq1 = load_frag(qb, HD, m0, 32, lane);

  const int off8 = (lane >> 4) << 3;
  const int q = lane & 15;
  const int qrow = m0 + q;      // this lane's query (duplicated across halves)
  const int sEnd = m0 + 15;

  float mi = -3.0e30f, li = 0.0f;
  v8f acc[4] = {};              // Oᵀ: lane q, rows d = 16j + off8 + r

  auto chunk = [&](auto maskc, int s0) {
    constexpr bool MASK = decltype(maskc)::value;
    // Sᵀ: A-fragments from K (rows = keys), K-dim = head dim.
    v16h a00 = load_frag(kb, HD, s0, 0, lane);
    v16h a01 = load_frag(kb, HD, s0, 32, lane);
    v16h a10 = load_frag(kb, HD, s0 + 16, 0, lane);
    v16h a11 = load_frag(kb, HD, s0 + 16, 32, lane);
    v8f t1 = {}, t2 = {};
    t1 = wmma_f16(a00, bq0, t1);
    t1 = wmma_f16(a01, bq1, t1);
    t2 = wmma_f16(a10, bq0, t2);
    t2 = wmma_f16(a11, bq1, t2);

    // in-lane chunk max; causal mask only on boundary chunks
    float cm = -3.0e30f;
    if constexpr (MASK) {
#pragma unroll
      for (int r = 0; r < 8; ++r) {
        const int k1 = s0 + r + off8;            // key of t1[r]
        const float v1 = (k1 <= qrow) ? (float)t1[r] : -3.0e30f;
        const float v2 = (k1 + 16 <= qrow) ? (float)t2[r] : -3.0e30f;
        t1[r] = v1;
        t2[r] = v2;
        cm = fmaxf(cm, fmaxf(v1, v2));
      }
    } else {
#pragma unroll
      for (int r = 0; r < 8; ++r) cm = fmaxf(cm, fmaxf(t1[r], t2[r]));
    }
    cm = fmaxf(cm, xhalf_swap(cm));  // combine the two lane halves
    const float mn = fmaxf(mi, cm);
    const float alpha = __expf(mi - mn);
    mi = mn;

    // exp + in-lane pack of Pᵀ into a ready-made B-fragment
    v16h up;
    float ps = 0.0f;
#pragma unroll
    for (int r = 0; r < 8; ++r) {
      const float p1 = __expf(t1[r] - mn);
      const float p2 = __expf(t2[r] - mn);
      ps += p1 + p2;
      up[r] = (_Float16)p1;       // keys kb..kb+7       (frag1)
      up[8 + r] = (_Float16)p2;   // keys 16+kb..16+kb+7 (frag2)
    }
    ps += xhalf_swap(ps);
    li = li * alpha + ps;

    // Oᵀ += Vᵀ·Pᵀ ; rescale is a scalar per lane (one query per lane)
#pragma unroll
    for (int j = 0; j < 4; ++j) {
      v16h av = load_frag(vb, TT, 16 * j, s0, lane);
      acc[j] *= alpha;
      acc[j] = wmma_f16(av, up, acc[j]);
    }
  };

  int s0 = 0;
  for (; s0 + 31 <= m0; s0 += 32)  // fully below the diagonal: no masking
    chunk(std::false_type{}, s0);
  for (; s0 <= sEnd; s0 += 32)     // diagonal boundary: causal mask
    chunk(std::true_type{}, s0);

  // epilogue: divide by li (in-lane), packed 16B stores (8 consecutive d)
  const float inv = 1.0f / li;
  _Float16* yb = yh + ((size_t)(b * TT + (m0 + q))) * DM + h * HD + off8;
#pragma unroll
  for (int j = 0; j < 4; ++j) {
    h8 pk;
#pragma unroll
    for (int r = 0; r < 8; ++r) pk[r] = (_Float16)(acc[j][r] * inv);
    *(h8*)(yb + 16 * j) = pk;
  }
}

extern "C" void kernel_launch(void* const* d_in, const int* in_sizes, int n_in,
                              void* d_out, int out_size, void* d_ws,
                              size_t ws_size, hipStream_t stream) {
  const float* x    = (const float*)d_in[0];
  const float* cosp = (const float*)d_in[1];
  const float* sinp = (const float*)d_in[2];
  const float* wq   = (const float*)d_in[3];
  const float* wk   = (const float*)d_in[4];
  const float* wv   = (const float*)d_in[5];
  const float* wo   = (const float*)d_in[6];

  char* ws = (char*)d_ws;
  size_t off = 0;
  auto alloc = [&](size_t bytes) {
    void* p = ws + off;
    off += (bytes + 255) & ~(size_t)255;
    return p;
  };
  const int M = BB * TT;                       // 4096
  _Float16* xh  = (_Float16*)alloc((size_t)M * DM * 2);
  _Float16* wqh = (_Float16*)alloc((size_t)DM * DM * 2);
  _Float16* wkh = (_Float16*)alloc((size_t)NKV * HD * DM * 2);
  _Float16* wvh = (_Float16*)alloc((size_t)NKV * HD * DM * 2);
  _Float16* woh = (_Float16*)alloc((size_t)DM * DM * 2);
  _Float16* qhp = (_Float16*)alloc((size_t)BB * NH * TT * HD * 2);
  _Float16* khp = (_Float16*)alloc((size_t)BB * NKV * TT * HD * 2);
  _Float16* vtp = (_Float16*)alloc((size_t)BB * NKV * HD * TT * 2);
  _Float16* yhp = (_Float16*)alloc((size_t)M * DM * 2);

  const int nx = M * DM, nq = DM * DM, nk = NKV * HD * DM;
  cvt_f32_f16<<<nx / 256, 256, 0, stream>>>(x, xh, nx);
  cvt_f32_f16<<<nq / 256, 256, 0, stream>>>(wq, wqh, nq);
  cvt_f32_f16<<<nk / 256, 256, 0, stream>>>(wk, wkh, nk);
  cvt_f32_f16<<<nk / 256, 256, 0, stream>>>(wv, wvh, nk);
  cvt_f32_f16<<<nq / 256, 256, 0, stream>>>(wo, woh, nq);

  dim3 blk(256);
  // Q projection + RoPE + 1/8 pre-scale
  gemm_wmma<1><<<dim3(DM / 128, M / 128), blk, 0, stream>>>(
      xh, wqh, nullptr, qhp, cosp, sinp, M, DM, DM);
  // K projection + RoPE
  gemm_wmma<2><<<dim3((NKV * HD) / 128, M / 128), blk, 0, stream>>>(
      xh, wkh, nullptr, khp, cosp, sinp, M, NKV * HD, DM);
  // V projection, transposed pack
  gemm_wmma<3><<<dim3((NKV * HD) / 128, M / 128), blk, 0, stream>>>(
      xh, wvh, nullptr, vtp, nullptr, nullptr, M, NKV * HD, DM);
  // causal flash attention (transposed formulation)
  attn_wmma<<<dim3(TT / 128, BB * NH), blk, 0, stream>>>(qhp, khp, vtp, yhp);
  // output projection, f32 out
  gemm_wmma<0><<<dim3(DM / 128, M / 128), blk, 0, stream>>>(
      yhp, woh, (float*)d_out, nullptr, nullptr, nullptr, M, DM, DM);
}
